// HierarchicalTimescaleController_53068615909653
// MI455X (gfx1250) — matrix operand order repaired and checked by reading, not compiled, not run on hardware
//
#include <hip/hip_runtime.h>
#include <hip/hip_bf16.h>
#include <math.h>

typedef __attribute__((ext_vector_type(16))) _Float16 v16h;
typedef __attribute__((ext_vector_type(8)))  float    v8f;

// ---------------------------------------------------------------------------
// helpers
// ---------------------------------------------------------------------------
__device__ __forceinline__ void pack8(v16h& v, int base, float4 a, float4 b) {
  v[base + 0] = (_Float16)a.x; v[base + 1] = (_Float16)a.y;
  v[base + 2] = (_Float16)a.z; v[base + 3] = (_Float16)a.w;
  v[base + 4] = (_Float16)b.x; v[base + 5] = (_Float16)b.y;
  v[base + 6] = (_Float16)b.z; v[base + 7] = (_Float16)b.w;
}

// ---------------------------------------------------------------------------
// WMMA GEMM:  C[M,N] (+)= A[M,K] @ W[N,K]^T  (+ bias[N])
// f32 in/out, f16 WMMA with f32 accumulate. One wave = 16x64 output strip.
// flags: bit0 = add bias, bit1 = accumulate into existing C
// M % 16 == 0, N % 64 == 0, K % 32 == 0 (guaranteed by problem shapes)
// ---------------------------------------------------------------------------
__global__ __launch_bounds__(256)
void gemm_wmma_f16(const float* __restrict__ A, int lda,
                   const float* __restrict__ W, int ldw,
                   const float* __restrict__ bias,
                   float* __restrict__ C, int ldc,
                   int M, int N, int K, int flags)
{
  int wave = (int)((blockIdx.x * blockDim.x + threadIdx.x) >> 5);
  int lane = threadIdx.x & 31;
  int nstrips = N >> 6;
  int mtiles  = M >> 4;
  if (wave >= mtiles * nstrips) return;
  int m0 = (wave / nstrips) << 4;
  int n0 = (wave % nstrips) << 6;
  int half   = lane >> 4;
  int lanelo = lane & 15;

  v8f acc[4];
#pragma unroll
  for (int j = 0; j < 4; ++j)
#pragma unroll
    for (int r = 0; r < 8; ++r) acc[j][r] = 0.f;

  // A-fragment: lane = row; K chunks [half*8, half*8+8) and [16+half*8, ...)
  const float* Arow = A + (size_t)(m0 + lanelo) * lda + half * 8;

  for (int k0 = 0; k0 < K; k0 += 32) {
    v16h a;
    {
      const float4* p0 = (const float4*)(Arow + k0);
      const float4* p1 = (const float4*)(Arow + k0 + 16);
      pack8(a, 0, p0[0], p0[1]);
      pack8(a, 8, p1[0], p1[1]);
    }
#pragma unroll
    for (int j = 0; j < 4; ++j) {
      // B-fragment: lane = column; K chunks [half*8, half*8+8) and [16+half*8, ...)
      const float* Wp = W + (size_t)(n0 + j * 16 + lanelo) * ldw + k0 + half * 8;
      const float4* q = (const float4*)Wp;
      v16h b;
      pack8(b, 0, q[0], q[1]);
      pack8(b, 8, q[4], q[5]);
      acc[j] = __builtin_amdgcn_wmma_f32_16x16x32_f16(
          false, a, false, b, (short)0, acc[j], false, false);
    }
  }

#pragma unroll
  for (int j = 0; j < 4; ++j) {
    int col  = n0 + j * 16 + lanelo;
    float bv = (flags & 1) ? bias[col] : 0.f;
#pragma unroll
    for (int r = 0; r < 8; ++r) {
      size_t idx = (size_t)(m0 + half * 8 + r) * ldc + col;
      float v = acc[j][r] + bv;
      if (flags & 2) v += C[idx];
      C[idx] = v;
    }
  }
}

// ---------------------------------------------------------------------------
// RMSNorm over d=512 with temporal subsampling: out[b,t,:] = rmsnorm(in[b,t*ts,:])*w
// one block (256 thr) per output row, 2 elements/thread
// ---------------------------------------------------------------------------
__global__ __launch_bounds__(256)
void rmsnorm_sub(const float* __restrict__ in, const float* __restrict__ w,
                 float* __restrict__ out, int T, int Sin, int ts)
{
  int row = blockIdx.x;                 // [0, B*T)
  int b = row / T, t = row - b * T;
  const float* src = in + ((size_t)b * Sin + (size_t)t * ts) * 512;
  float* dst = out + (size_t)row * 512;
  int tid = threadIdx.x;
  float v0 = src[tid], v1 = src[tid + 256];
  float ss = v0 * v0 + v1 * v1;
#pragma unroll
  for (int off = 16; off > 0; off >>= 1) ss += __shfl_xor(ss, off, 32);
  __shared__ float red[8];
  if ((tid & 31) == 0) red[tid >> 5] = ss;
  __syncthreads();
  float tot = 0.f;
#pragma unroll
  for (int k = 0; k < 8; ++k) tot += red[k];
  float scale = rsqrtf(tot * (1.f / 512.f) + 1e-6f);
  dst[tid]       = v0 * scale * w[tid];
  dst[tid + 256] = v1 * scale * w[tid + 256];
}

// final: out = rmsnorm(comb)*w + residual(x); one block per row
__global__ __launch_bounds__(256)
void final_norm(const float* __restrict__ comb, const float* __restrict__ w,
                const float* __restrict__ x, float* __restrict__ out)
{
  int row = blockIdx.x;
  const float* src = comb + (size_t)row * 512;
  const float* res = x    + (size_t)row * 512;
  float* dst = out + (size_t)row * 512;
  int tid = threadIdx.x;
  float v0 = src[tid], v1 = src[tid + 256];
  float ss = v0 * v0 + v1 * v1;
#pragma unroll
  for (int off = 16; off > 0; off >>= 1) ss += __shfl_xor(ss, off, 32);
  __shared__ float red[8];
  if ((tid & 31) == 0) red[tid >> 5] = ss;
  __syncthreads();
  float tot = 0.f;
#pragma unroll
  for (int k = 0; k < 8; ++k) tot += red[k];
  float scale = rsqrtf(tot * (1.f / 512.f) + 1e-6f);
  dst[tid]       = v0 * scale * w[tid]       + res[tid];
  dst[tid + 256] = v1 * scale * w[tid + 256] + res[tid + 256];
}

// ---------------------------------------------------------------------------
// c_sum[i,s] = sum_d Cm[i,d,s]
// ---------------------------------------------------------------------------
__global__ void csum_kernel(const float* __restrict__ Cm, float* __restrict__ csum)
{
  int idx = blockIdx.x * blockDim.x + threadIdx.x;
  if (idx >= 3 * 64) return;
  int i = idx >> 6, s = idx & 63;
  const float* p = Cm + (size_t)i * 512 * 64 + s;
  float acc = 0.f;
  for (int dd = 0; dd < 512; ++dd) acc += p[dd * 64];
  csum[idx] = acc;
}

// ---------------------------------------------------------------------------
// Sequential SSM scan. One wave per (b, d) channel; 2 states per lane (64).
// xz rows are (B*T, 1024); x_proj = cols [0,512). ys: (B*T, 512).
// state = tanh(state*A + x_t*Bm); y = sum_s state*c_sum + Dv*x_t
// B=2, d=512 hardcoded (problem shape).
// ---------------------------------------------------------------------------
__global__ __launch_bounds__(256)
void scan_kernel(const float* __restrict__ xz,
                 const float* __restrict__ Ai, const float* __restrict__ Bi,
                 const float* __restrict__ cs, const float* __restrict__ Dvi,
                 float* __restrict__ ys, int T)
{
  int lane = threadIdx.x & 31;
  int row  = blockIdx.x * 8 + (threadIdx.x >> 5);   // [0, 1024)
  if (row >= 1024) return;
  int b = row >> 9, dch = row & 511;
  float a0  = Ai[dch * 64 + lane], a1  = Ai[dch * 64 + lane + 32];
  float bb0 = Bi[dch * 64 + lane], bb1 = Bi[dch * 64 + lane + 32];
  float c0  = cs[lane],            c1  = cs[lane + 32];
  float Dd  = Dvi[dch];
  float s0 = 0.f, s1 = 0.f;
  const float* xp = xz + (size_t)b * T * 1024 + dch;
  float*       yp = ys + (size_t)b * T * 512  + dch;
  for (int t = 0; t < T; ++t) {
    float xt = xp[(size_t)t * 1024];
    s0 = tanhf(fmaf(xt, bb0, s0 * a0));
    s1 = tanhf(fmaf(xt, bb1, s1 * a1));
    float part = fmaf(s0, c0, s1 * c1);
#pragma unroll
    for (int off = 16; off > 0; off >>= 1) part += __shfl_xor(part, off, 32);
    if (lane == 0) yp[(size_t)t * 512] = fmaf(Dd, xt, part);
  }
}

// u = ys * silu(gate),  gate = xz[:, 512 + c]
__global__ void silu_mul(const float* __restrict__ ys, const float* __restrict__ xz,
                         float* __restrict__ u, int n)
{
  int i = blockIdx.x * blockDim.x + threadIdx.x;
  if (i >= n) return;
  int r = i >> 9, c = i & 511;
  float g = xz[(size_t)r * 1024 + 512 + c];
  float sg = g / (1.f + expf(-g));
  u[i] = ys[i] * sg;
}

// nearest-neighbor temporal upsample: full[b,s,:] = sub[b, s/ts, :]
__global__ void upsample(const float* __restrict__ sub, float* __restrict__ full,
                         int T, int ts, int n)
{
  int i = blockIdx.x * blockDim.x + threadIdx.x;
  if (i >= n) return;
  int c = i & 511;
  int rest = i >> 9;
  int s = rest & 4095;
  int b = rest >> 12;
  full[i] = sub[(((size_t)b * T) + s / ts) * 512 + c];
}

// g = sigmoid(gpre); fast' = g*f + (1-g)*s; slow' = (1-g)*f + g*s  (in place)
__global__ void modmix(float* __restrict__ fast, float* __restrict__ slow,
                       const float* __restrict__ gpre, int n)
{
  int i = blockIdx.x * blockDim.x + threadIdx.x;
  if (i >= n) return;
  float g  = 1.f / (1.f + expf(-gpre[i]));
  float f  = fast[i], sl = slow[i];
  fast[i] = g * f + (1.f - g) * sl;
  slow[i] = (1.f - g) * f + g * sl;
}

// ---------------------------------------------------------------------------
// host side
// ---------------------------------------------------------------------------
static inline void launch_gemm(const float* A, int lda, const float* W, int ldw,
                               const float* bias, float* C, int ldc,
                               int M, int N, int K, int flags, hipStream_t stream)
{
  int waves  = (M >> 4) * (N >> 6);
  int blocks = (waves + 7) / 8;
  gemm_wmma_f16<<<blocks, 256, 0, stream>>>(A, lda, W, ldw, bias, C, ldc, M, N, K, flags);
}

extern "C" void kernel_launch(void* const* d_in, const int* in_sizes, int n_in,
                              void* d_out, int out_size, void* d_ws, size_t ws_size,
                              hipStream_t stream)
{
  const float* x        = (const float*)d_in[0];
  const float* in_norm  = (const float*)d_in[1];
  const float* ssm_norm = (const float*)d_in[2];
  const float* Win      = (const float*)d_in[3];
  const float* b_in     = (const float*)d_in[4];
  const float* Wout     = (const float*)d_in[5];
  const float* b_out    = (const float*)d_in[6];
  const float* Amat     = (const float*)d_in[7];
  const float* Bmat     = (const float*)d_in[8];
  const float* Cmat     = (const float*)d_in[9];
  const float* Dv       = (const float*)d_in[10];
  const float* gate_W   = (const float*)d_in[11];
  const float* gate_b   = (const float*)d_in[12];
  const float* comb_W   = (const float*)d_in[13];
  const float* comb_b   = (const float*)d_in[14];
  const float* out_norm = (const float*)d_in[15];

  const int Bsz = 2, S = 4096, d = 512;
  const size_t BSd = (size_t)Bsz * S * d;

  float* ws    = (float*)d_ws;
  float* h     = ws;               // (B,S,d)
  float* outs0 = ws + 1 * BSd;     // upsampled per-scale outputs
  float* outs1 = ws + 2 * BSd;
  float* outs2 = ws + 3 * BSd;
  float* xsub  = ws + 4 * BSd;     // also: u, gate-preactivation
  float* ys    = ws + 5 * BSd;     // also: combined (pre-norm)
  float* xz    = ws + 6 * BSd;     // (B*T, 2d) — 2*BSd
  float* csum  = ws + 8 * BSd;     // 3*64 floats
  float* outs[3] = {outs0, outs1, outs2};

  // C-sum reduction + input rmsnorm
  csum_kernel<<<1, 256, 0, stream>>>(Cmat, csum);
  rmsnorm_sub<<<Bsz * S, 256, 0, stream>>>(x, in_norm, h, S, S, 1);

  const int tss[3] = {1, 4, 16};
  for (int i = 0; i < 3; ++i) {
    int ts = tss[i], T = S / ts, M = Bsz * T;
    // subsample + per-scale rmsnorm
    rmsnorm_sub<<<M, 256, 0, stream>>>(h, ssm_norm + i * d, xsub, T, S, ts);
    // xz = xsub @ Win^T + b_in   (M x 1024, K=512)
    launch_gemm(xsub, d, Win + (size_t)i * 2 * d * d, d, b_in + i * 2 * d,
                xz, 2 * d, M, 2 * d, d, 1, stream);
    // sequential nonlinear scan
    scan_kernel<<<(Bsz * d) / 8, 256, 0, stream>>>(
        xz, Amat + (size_t)i * d * 64, Bmat + (size_t)i * d * 64,
        csum + i * 64, Dv + i * d, ys, T);
    // u = ys * silu(gate)
    int n1 = M * d;
    silu_mul<<<(n1 + 255) / 256, 256, 0, stream>>>(ys, xz, xsub, n1);
    // out_sub = u @ Wout^T + b_out  (reuse ys)
    launch_gemm(xsub, d, Wout + (size_t)i * d * d, d, b_out + i * d,
                ys, d, M, d, d, 1, stream);
    // upsample to full S
    int n2 = Bsz * S * d;
    upsample<<<(n2 + 255) / 256, 256, 0, stream>>>(ys, outs[i], T, ts, n2);
  }

  // slow -> fast modulation (i = 1, then 0); gate input = concat(fast, slow)
  for (int i = 1; i >= 0; --i) {
    // gpre  = fast @ gate_W[i][:, :d]^T
    launch_gemm(outs[i], d, gate_W + (size_t)i * d * 2 * d, 2 * d, nullptr,
                xsub, d, Bsz * S, d, d, 0, stream);
    // gpre += slow @ gate_W[i][:, d:]^T + gate_b[i]
    launch_gemm(outs[i + 1], d, gate_W + (size_t)i * d * 2 * d + d, 2 * d,
                gate_b + i * d, xsub, d, Bsz * S, d, d, 3, stream);
    int n = Bsz * S * d;
    modmix<<<(n + 255) / 256, 256, 0, stream>>>(outs[i], outs[i + 1], xsub, n);
  }

  // combined @ combine_W^T + b  as 3 K-split GEMMs into ys
  launch_gemm(outs0, d, comb_W,         3 * d, nullptr, ys, d, Bsz * S, d, d, 0, stream);
  launch_gemm(outs1, d, comb_W + d,     3 * d, nullptr, ys, d, Bsz * S, d, d, 2, stream);
  launch_gemm(outs2, d, comb_W + 2 * d, 3 * d, comb_b,  ys, d, Bsz * S, d, d, 3, stream);

  // out = rmsnorm(ys) * out_norm_w + residual
  final_norm<<<Bsz * S, 256, 0, stream>>>(ys, out_norm, x, (float*)d_out);
}
